// InvEig_42838003810954
// MI455X (gfx1250) — compile-verified, matrix operand order checked by reading, not run on hardware
//
#include <hip/hip_runtime.h>
#include <hip/hip_bf16.h>

typedef __attribute__((ext_vector_type(16))) _Float16 v16h;
typedef __attribute__((ext_vector_type(8)))  float    v8f;

// ---------------------------------------------------------------------------
// GEMV via WMMA with two-term f16 precision split (~fp32 accuracy):
//   y[j] = act( sum_k x[k] * W[k][j] + bias[j] ),  W row-major [K x N].
// One wave computes 16 output columns; the input vector is replicated across
// all 16 rows of the A fragment so every row of the 16x16 D tile equals the
// GEMV result (row 0 is written out).
//
// Precision: x = xh + xl, w = wh + wl (f16 hi/lo split);
//   x*w ~= xh*wh + xh*wl + xl*wh  (three chained WMMAs, error ~2^-22/term).
// Compute is free here: the kernel is bandwidth-bound on the fp32 weight
// stream (58 MB total / 23.3 TB/s ~= 2.5 us floor).
//
// 16-bit A/B fragment K-permutation (CDNA5 ISA 7.12.2, 16-bit 16x32 layout):
//   element e (0..15) in lane-half h holds K = e + (e & 8) + 8*h
// ---------------------------------------------------------------------------
__global__ void gemv_wmma_kernel(const float* __restrict__ x,
                                 const float* __restrict__ W,
                                 const float* __restrict__ bias,
                                 float* __restrict__ y,
                                 int K, int N, int do_relu) {
    extern __shared__ float sx[];                 // K floats: staged input vec
    const int tid  = threadIdx.x;
    const int lane = tid & 31;
    const int wave = tid >> 5;

    for (int i = tid; i < K; i += blockDim.x) sx[i] = x[i];
    __syncthreads();

    const int j0  = blockIdx.x * 64 + wave * 16;  // 4 waves x 16 cols = 64/block
    const int col = j0 + (lane & 15);
    const int hi8 = (lane >> 4) * 8;              // K offset for upper lane half

    v8f acc = {};
    for (int k0 = 0; k0 < K; k0 += 32) {
        v16h ah = {}, al = {}, bh = {}, bl = {};
#pragma unroll
        for (int e = 0; e < 16; ++e) {
            const int kk = k0 + e + (e & 8) + hi8;          // K permutation
            const float xv = sx[kk];                        // LDS broadcast
            const float wv = W[(long)kk * N + col];         // coalesced 64B/row
            const _Float16 xh = (_Float16)xv;
            const _Float16 wh = (_Float16)wv;
            ah[e] = xh;  al[e] = (_Float16)(xv - (float)xh);
            bh[e] = wh;  bl[e] = (_Float16)(wv - (float)wh);
        }
        // D = A(16x32 f16) * B(32x16 f16) + C(f32), three-term split product
        acc = __builtin_amdgcn_wmma_f32_16x16x32_f16(
            false, ah, false, bh, (short)0, acc, false, false);
        acc = __builtin_amdgcn_wmma_f32_16x16x32_f16(
            false, ah, false, bl, (short)0, acc, false, false);
        acc = __builtin_amdgcn_wmma_f32_16x16x32_f16(
            false, al, false, bh, (short)0, acc, false, false);
    }

    if (lane < 16) {                              // D row M=0 lives in VGPR0, lanes 0..15
        float v = acc[0] + bias[col];
        if (do_relu) v = fmaxf(v, 0.0f);
        y[col] = v;
    }
}

// ---------------------------------------------------------------------------
// Eigenvalues of symmetric tridiagonal H (off-diag = -1/2, e^2 = 1/4) by
// Sturm-sequence bisection.  blockIdx.y = l (0/1); one thread per eigenvalue
// index.  Diagonal staged in LDS (16 KB of 320 KB/WGP); inner recurrence uses
// v_rcp_f32 and an LDS-broadcast read -- ~48 x 4096 dependent steps per lane,
// latency hidden across 8192 concurrent eigenvalue searches.
// ---------------------------------------------------------------------------
#define RN 4096

__global__ void eig_bisect_kernel(const float* __restrict__ ptl,
                                  float* __restrict__ evl) {
    __shared__ float sd[RN];
    const int   l   = blockIdx.y;
    const float ll1 = (float)(l * (l + 1));
    const int   tid = threadIdx.x;
    const float dr  = 10000.0f / (float)RN;       // R_MAX / rn

    for (int i = tid; i < RN; i += blockDim.x) {
        const float r = dr * (float)(i + 1);
        sd[i] = 1.0f + ptl[i] + ll1 / (r * r);
    }
    __syncthreads();

    // Gershgorin bounds (one LDS scan ~= one bisection iteration's cost)
    float lo = 1e30f, hi = -1e30f;
    for (int i = 0; i < RN; ++i) {
        const float d = sd[i];
        lo = fminf(lo, d - 1.0f);
        hi = fmaxf(hi, d + 1.0f);
    }

    const int j = blockIdx.x * blockDim.x + tid;  // ascending eigenvalue index

    for (int it = 0; it < 48; ++it) {
        const float x = 0.5f * (lo + hi);
        // Sturm count: #negative leading-minor ratios = #eigenvalues < x
        float q   = sd[0] - x;
        int   cnt = (q < 0.0f);
        for (int i = 1; i < RN; ++i) {
            q = (sd[i] - x) - 0.25f * __builtin_amdgcn_rcpf(q);
            if (fabsf(q) < 1e-30f) q = -1e-30f;   // guard against breakdown
            cnt += (q < 0.0f);
        }
        if (cnt > j) hi = x; else lo = x;
    }
    evl[l * RN + j] = 0.5f * (lo + hi);
}

// ---------------------------------------------------------------------------
extern "C" void kernel_launch(void* const* d_in, const int* in_sizes, int n_in,
                              void* d_out, int out_size, void* d_ws, size_t ws_size,
                              hipStream_t stream) {
    const float* energy = (const float*)d_in[0];  // [4096]
    const float* W1     = (const float*)d_in[1];  // [4096,2048]
    const float* b1     = (const float*)d_in[2];  // [2048]
    const float* W2     = (const float*)d_in[3];  // [2048,1024]
    const float* b2     = (const float*)d_in[4];  // [1024]
    const float* W3     = (const float*)d_in[5];  // [1024,4096]
    const float* b3     = (const float*)d_in[6];  // [4096]

    float* out = (float*)d_out;                   // [4096 ptl | 2*4096 evl]
    float* h1  = (float*)d_ws;                    // [2048]
    float* h2  = h1 + 2048;                       // [1024]

    // h1 = relu(energy @ W1 + b1)
    gemv_wmma_kernel<<<2048 / 64, 128, 4096 * sizeof(float), stream>>>(
        energy, W1, b1, h1, 4096, 2048, 1);
    // h2 = relu(h1 @ W2 + b2)
    gemv_wmma_kernel<<<1024 / 64, 128, 2048 * sizeof(float), stream>>>(
        h1, W2, b2, h2, 2048, 1024, 1);
    // ptl = h2 @ W3 + b3  -> d_out[0:4096]
    gemv_wmma_kernel<<<4096 / 64, 128, 1024 * sizeof(float), stream>>>(
        h2, W3, b3, out, 1024, 4096, 0);
    // evl[l, j] -> d_out[4096 + l*4096 + j]
    eig_bisect_kernel<<<dim3(RN / 256, 2), 256, 0, stream>>>(out, out + 4096);
}
